// BinaryConv_39444979646683
// MI455X (gfx1250) — 8-bit matrix operand order repaired and checked by reading, compiled, not run on hardware
//
#include <hip/hip_runtime.h>

typedef int v8i   __attribute__((ext_vector_type(8)));
typedef int v4i_b __attribute__((vector_size(16)));

#define NB   32
#define CIN  256
#define COUT 256
#define HW   1024
#define MTOT (NB*HW)
#define KSTEPS 36
#define HP   34

#define ACTP_BYTES (NB*HP*HP*CIN)
#define WBP_BYTES  (9*4*COUT*64)
#define OFF_ACTP   0
#define OFF_WBP    (OFF_ACTP + ACTP_BYTES)
#define OFF_SCALE  (OFF_WBP + WBP_BYTES)
#define OFF_SHIFT  (OFF_SCALE + 1024)

#if defined(__has_builtin)
#  if __has_builtin(__builtin_amdgcn_global_load_async_to_lds_b128) && \
      __has_builtin(__builtin_amdgcn_s_wait_asynccnt)
#    define HAVE_ASYNC_LDS 1
#  endif
#endif
#ifndef HAVE_ASYNC_LDS
#  define HAVE_ASYNC_LDS 0
#endif

#define GLOBAL_V4(p) ((__attribute__((address_space(1))) v4i_b*)(p))
#define LDS_V4(p)    ((__attribute__((address_space(3))) v4i_b*)(p))

__global__ __launch_bounds__(256)
void bn_stats(const float* __restrict__ x,
              const float* __restrict__ gamma,
              const float* __restrict__ beta,
              float* __restrict__ scale,
              float* __restrict__ shift) {
    __shared__ float s_sum[256];
    __shared__ float s_sq[256];
    const int c = blockIdx.x;
    const int t = threadIdx.x;
    float sum = 0.f, sq = 0.f;
    for (int j = t; j < MTOT; j += 256) {
        const int n = j >> 10, hw = j & 1023;
        const float v = x[((n * CIN + c) << 10) + hw];
        sum += v; sq += v * v;
    }
    s_sum[t] = sum; s_sq[t] = sq;
    __syncthreads();
    for (int s = 128; s > 0; s >>= 1) {
        if (t < s) { s_sum[t] += s_sum[t + s]; s_sq[t] += s_sq[t + s]; }
        __syncthreads();
    }
    if (t == 0) {
        const float mean = s_sum[0] * (1.0f / MTOT);
        const float var  = s_sq[0] * (1.0f / MTOT) - mean * mean;
        const float sc   = gamma[c] * rsqrtf(var + 1e-5f);
        scale[c] = sc;
        shift[c] = beta[c] - mean * sc;
    }
}

__global__ __launch_bounds__(256)
void binarize_act_pad(const float* __restrict__ x,
                      const float* __restrict__ scale,
                      const float* __restrict__ shift,
                      signed char* __restrict__ actp) {
    const int idx = blockIdx.x * 256 + threadIdx.x;
    const int c   = idx & 255;
    const int pix = idx >> 8;
    const int n   = pix >> 10, hw = pix & 1023;
    const int h   = hw >> 5,   w  = hw & 31;
    const float v = x[((n * CIN + c) << 10) + hw] * scale[c] + shift[c];
    actp[((n * HP + (h + 1)) * HP + (w + 1)) * 256 + c] =
        (signed char)(v > 0.f ? 1 : (v < 0.f ? -1 : 0));
}

__global__ __launch_bounds__(256)
void binarize_wt(const float* __restrict__ Wt,
                 signed char* __restrict__ wbp) {
    const int idx  = blockIdx.x * 256 + threadIdx.x;
    const int k    = idx & 63;
    const int ko   = (idx >> 6) & 255;
    const int cblk = (idx >> 14) & 3;
    const int tap  = idx >> 16;
    const int ci   = (cblk << 6) + k;
    const int kh   = tap / 3, kw = tap % 3;
    const float v  = Wt[((ko * CIN + ci) * 3 + kh) * 3 + kw];
    wbp[idx] = (signed char)(v > 0.f ? 1 : (v < 0.f ? -1 : 0));
}

__global__ __launch_bounds__(256)
void bgemm_wmma(const signed char* __restrict__ actp,
                const signed char* __restrict__ wbp,
                float* __restrict__ out) {
    __shared__ signed char la[2][128 * 64];
    __shared__ signed char lb[2][64 * 64];

    const int t      = threadIdx.x;
    const int mBase  = blockIdx.x * 128;
    const int nBase  = blockIdx.y * 64;
    const int wave   = t >> 5;
    const int lane   = t & 31;
    const int laneHi = lane >> 4;
    const int laneLo = lane & 15;

    v8i acc0 = {}, acc1 = {}, acc2 = {}, acc3 = {};

    const int ar = t >> 1;
    const int ah = t & 1;
    const int m     = mBase + ar;
    const int n_img = m >> 10;
    const int h     = (m >> 5) & 31;
    const int w     = m & 31;
    const signed char* aThreadBase =
        actp + (((n_img * HP + h) * HP + w) << 8) + (ah << 5);
    signed char* aDstBase = &la[0][ar * 64 + (ah << 5)];

    const int bc = t >> 2;
    const int bq = t & 3;
    const signed char* bThreadBase = wbp + ((nBase + bc) << 6) + (bq << 4);
    signed char* bDstBase = &lb[0][bc * 64 + (bq << 4)];

    const int ldsAStride = 128 * 64;
    const int ldsBStride = 64 * 64;

    auto stage = [&](int buf, int ks) {
        const int tap  = ks >> 2;
        const int cblk = ks & 3;
        const int kh   = tap / 3;
        const int kw   = tap - kh * 3;
        const signed char* aSrc = aThreadBase + ((kh * HP + kw) << 8) + (cblk << 6);
        const signed char* bSrc = bThreadBase + ks * (256 * 64);
        signed char* aDst = aDstBase + buf * ldsAStride;
        signed char* bDst = bDstBase + buf * ldsBStride;
#if HAVE_ASYNC_LDS
        __builtin_amdgcn_global_load_async_to_lds_b128(GLOBAL_V4(aSrc),      LDS_V4(aDst),      0, 0);
        __builtin_amdgcn_global_load_async_to_lds_b128(GLOBAL_V4(aSrc + 16), LDS_V4(aDst + 16), 0, 0);
        __builtin_amdgcn_global_load_async_to_lds_b128(GLOBAL_V4(bSrc),      LDS_V4(bDst),      0, 0);
#else
        const int4 p0 = ((const int4*)aSrc)[0];
        const int4 p1 = ((const int4*)aSrc)[1];
        const int4 q0 = ((const int4*)bSrc)[0];
        ((int4*)aDst)[0] = p0;
        ((int4*)aDst)[1] = p1;
        ((int4*)bDst)[0] = q0;
#endif
    };

    stage(0, 0);

    for (int ks = 0; ks < KSTEPS; ++ks) {
        const int buf = ks & 1;
        if (ks + 1 < KSTEPS) {
            stage(buf ^ 1, ks + 1);
#if HAVE_ASYNC_LDS
            __builtin_amdgcn_s_wait_asynccnt(3);
#endif
        } else {
#if HAVE_ASYNC_LDS
            __builtin_amdgcn_s_wait_asynccnt(0);
#endif
        }
        __syncthreads();

        v8i A;
        {
            const int row = (wave << 4) + laneLo;
            const int4* ap = (const int4*)(&la[buf][row * 64 + (laneHi << 4)]);
            const int4 q0 = ap[0], q1 = ap[2];
            A[0] = q0.x; A[1] = q0.y; A[2] = q0.z; A[3] = q0.w;
            A[4] = q1.x; A[5] = q1.y; A[6] = q1.z; A[7] = q1.w;
        }
#define BFRAG_MMA(CB, ACC)                                                     \
        {                                                                      \
            const int col = ((CB) << 4) + laneLo;                              \
            const int4* bp = (const int4*)(&lb[buf][col * 64 + (laneHi << 4)]);\
            const int4 b0 = bp[0], b1 = bp[2];                                 \
            v8i B;                                                             \
            B[0] = b0.x; B[1] = b0.y; B[2] = b0.z; B[3] = b0.w;                \
            B[4] = b1.x; B[5] = b1.y; B[6] = b1.z; B[7] = b1.w;                \
            ACC = __builtin_amdgcn_wmma_i32_16x16x64_iu8(                      \
                      true, A, true, B, ACC, false, false);                    \
        }
        BFRAG_MMA(0, acc0)
        BFRAG_MMA(1, acc1)
        BFRAG_MMA(2, acc2)
        BFRAG_MMA(3, acc3)
#undef BFRAG_MMA
        __syncthreads();
    }

#define STORE_ACC(CB, ACC)                                                     \
    {                                                                          \
        const int col = nBase + ((CB) << 4) + laneLo;                          \
        _Pragma("unroll")                                                      \
        for (int j = 0; j < 8; ++j) {                                          \
            const int mrow = mBase + (wave << 4) + (laneHi << 3) + j;          \
            out[(((mrow >> 10) * COUT + col) << 10) + (mrow & 1023)] =         \
                (float)ACC[j];                                                 \
        }                                                                      \
    }
    STORE_ACC(0, acc0)
    STORE_ACC(1, acc1)
    STORE_ACC(2, acc2)
    STORE_ACC(3, acc3)
#undef STORE_ACC
}

extern "C" void kernel_launch(void* const* d_in, const int* in_sizes, int n_in,
                              void* d_out, int out_size, void* d_ws, size_t ws_size,
                              hipStream_t stream) {
    const float* x     = (const float*)d_in[0];
    const float* gamma = (const float*)d_in[1];
    const float* beta  = (const float*)d_in[2];
    const float* W     = (const float*)d_in[3];
    float* out = (float*)d_out;

    signed char* actp  = (signed char*)d_ws + OFF_ACTP;
    signed char* wbp   = (signed char*)d_ws + OFF_WBP;
    float*       scale = (float*)((char*)d_ws + OFF_SCALE);
    float*       shift = (float*)((char*)d_ws + OFF_SHIFT);

    (void)hipMemsetAsync(actp, 0, ACTP_BYTES, stream);

    bn_stats<<<CIN, 256, 0, stream>>>(x, gamma, beta, scale, shift);

    binarize_act_pad<<<(MTOT * CIN) / 256, 256, 0, stream>>>(x, scale, shift, actp);

    binarize_wt<<<WBP_BYTES / 256, 256, 0, stream>>>(W, wbp);

    dim3 grid(MTOT / 128, COUT / 64, 1);
    bgemm_wmma<<<grid, 256, 0, stream>>>(actp, wbp, out);
}
